// SkeletonizingAndGroupingLayer_75600014344431
// MI455X (gfx1250) — compile-verified, matrix operand order checked by reading, not run on hardware
//
#include <hip/hip_runtime.h>
#include <hip/hip_bf16.h>

// ---------------- problem constants (from reference) ----------------
#define BB   8
#define NN   8192
#define CC   64
#define MM   2048          // NUM_FPS
#define KNB  32            // K neighbors
#define D1P  96            // padded K-dim of GEMM1: [rel(3) | 0 | feat(64) | 0 x28]
#define HD   128           // hidden dim
#define BM   (BB*MM)       // 16384 centers total

typedef __attribute__((ext_vector_type(16))) _Float16 v16h;
typedef __attribute__((ext_vector_type(8)))  _Float16 v8h;
typedef __attribute__((ext_vector_type(4)))  _Float16 v4h;
typedef __attribute__((ext_vector_type(8)))  float    v8f;

__device__ __forceinline__ v16h make_v16h(v8h lo, v8h hi) {
    v16h r;
#pragma unroll
    for (int i = 0; i < 8; ++i) { r[i] = lo[i]; r[i + 8] = hi[i]; }
    return r;
}

// =====================================================================
// Kernel 0: convert W1/W2 (f32, row-major KxN) into f16 blobs laid out in
// the exact per-lane WMMA B-fragment order (ISA 7.12.2, 16-bit 16x32):
//   halfword index = ((kt*8+nt)*512) + lane*16 + j
//   K = kt*32 + (lane>=16 ? 8 : 0) + (j<8 ? j : j+8)   (j+8 -> K+16..K+23)
//   N = nt*16 + (lane&15)
// W1 K-rows are remapped to the padded gather layout:
//   lds-col 0..2  -> W1 row 0..2   (rel embed)
//   lds-col 3     -> zero          (alignment pad)
//   lds-col 4..67 -> W1 row 3..66  (features)
//   lds-col 68..95-> zero          (K padding to 96)
// =====================================================================
__global__ void prep_weights_kernel(const float* __restrict__ W1,
                                    const float* __restrict__ W2,
                                    _Float16* __restrict__ w1f,
                                    _Float16* __restrict__ w2f) {
    int t = blockIdx.x * 256 + threadIdx.x;          // 28672 total halves
    if (t < 3 * 8 * 512) {                           // W1: 3 k-tiles x 8 n-tiles
        int frag = t >> 9, within = t & 511;
        int lane = within >> 4, j = within & 15;
        int kt = frag >> 3, nt = frag & 7;
        int Kk = kt * 32 + ((lane >= 16) ? 8 : 0) + ((j < 8) ? j : j + 8);
        int Nn = nt * 16 + (lane & 15);
        float val;
        if (Kk < 3)        val = W1[Kk * HD + Nn];          // rel embed rows
        else if (Kk == 3)  val = 0.f;                       // pad column
        else if (Kk < 68)  val = W1[(Kk - 1) * HD + Nn];    // feature rows 3..66
        else               val = 0.f;                       // K padding
        w1f[t] = (_Float16)val;
    } else if (t < 3 * 8 * 512 + 4 * 8 * 512) {      // W2: 4 k-tiles x 8 n-tiles
        int t2 = t - 3 * 8 * 512;
        int frag = t2 >> 9, within = t2 & 511;
        int lane = within >> 4, j = within & 15;
        int kt = frag >> 3, nt = frag & 7;
        int Kk = kt * 32 + ((lane >= 16) ? 8 : 0) + ((j < 8) ? j : j + 8);
        int Nn = nt * 16 + (lane & 15);
        w2f[t2] = (_Float16)W2[Kk * HD + Nn];
    }
}

// =====================================================================
// Kernel 1: furthest point sampling. One workgroup (256 thr = 8 waves)
// per batch. Points + running min-distances live in registers
// (32/thread). Per-iteration argmax: shfl_xor wave reduce, then LDS
// across 8 waves. 2047 sequential iterations (latency-bound by design).
// =====================================================================
__global__ __launch_bounds__(256) void fps_kernel(const float* __restrict__ xyz,
                                                  int* __restrict__ sids) {
    __shared__ float swd[8];
    __shared__ int   swi[8];
    __shared__ int   sbcast;
    const int b   = blockIdx.x;
    const int tid = threadIdx.x;
    const float* X = xyz + (size_t)b * NN * 3;

    float px[32], py[32], pz[32], dd[32];
#pragma unroll
    for (int i = 0; i < 32; ++i) {
        int p = tid + i * 256;
        px[i] = X[p * 3 + 0]; py[i] = X[p * 3 + 1]; pz[i] = X[p * 3 + 2];
        dd[i] = 1e10f;
    }
    if (tid == 0) sids[b * MM] = 0;

    int last = 0;
    for (int t = 1; t < MM; ++t) {
        float lx = X[last * 3 + 0], ly = X[last * 3 + 1], lz = X[last * 3 + 2];
        float bd = -1.0f; int bi = 0;
#pragma unroll
        for (int i = 0; i < 32; ++i) {
            float dx = px[i] - lx, dy = py[i] - ly, dz = pz[i] - lz;
            float d = dx * dx + dy * dy + dz * dz;
            dd[i] = fminf(dd[i], d);
            int p = tid + i * 256;
            if (dd[i] > bd || (dd[i] == bd && p < bi)) { bd = dd[i]; bi = p; }
        }
        // wave32 argmax (first-index tie-break like jnp.argmax)
        for (int off = 16; off > 0; off >>= 1) {
            float od = __shfl_xor(bd, off, 32);
            int   oi = __shfl_xor(bi, off, 32);
            if (od > bd || (od == bd && oi < bi)) { bd = od; bi = oi; }
        }
        if ((tid & 31) == 0) { swd[tid >> 5] = bd; swi[tid >> 5] = bi; }
        __syncthreads();
        if (tid == 0) {
            float gb = swd[0]; int gi = swi[0];
            for (int k = 1; k < 8; ++k)
                if (swd[k] > gb || (swd[k] == gb && swi[k] < gi)) { gb = swd[k]; gi = swi[k]; }
            sids[b * MM + t] = gi;
            sbcast = gi;
        }
        __syncthreads();
        last = sbcast;
    }
}

// =====================================================================
// Kernel 2: gather centers / center_embed, emit sample_ids as float.
// =====================================================================
__global__ void gather_centers_kernel(const float* __restrict__ xyz,
                                      const float* __restrict__ xyz_embed,
                                      const int* __restrict__ sids,
                                      float* __restrict__ out_centers,
                                      float* __restrict__ out_cembed,
                                      float* __restrict__ out_ids) {
    int i = blockIdx.x * 256 + threadIdx.x;
    if (i >= BM) return;
    int b = i >> 11;
    int sid = sids[i];
    const float* p = xyz       + ((size_t)b * NN + sid) * 3;
    const float* e = xyz_embed + ((size_t)b * NN + sid) * 3;
    out_centers[i * 3 + 0] = p[0]; out_centers[i * 3 + 1] = p[1]; out_centers[i * 3 + 2] = p[2];
    out_cembed[i * 3 + 0]  = e[0]; out_cembed[i * 3 + 1]  = e[1]; out_cembed[i * 3 + 2]  = e[2];
    out_ids[i] = (float)sid;
}

// =====================================================================
// Kernel 3: KNN. One wave per center: d^2 to all 8192 points into 32KB
// LDS, then 32 rounds of wave-argmin with masking (xyz is L2-resident).
// =====================================================================
__global__ __launch_bounds__(32) void knn_kernel(const float* __restrict__ xyz,
                                                 const float* __restrict__ centers,
                                                 int* __restrict__ nn_idx) {
    __shared__ float dist[NN];
    const int c = blockIdx.x;
    const int b = c >> 11;
    const int lane = threadIdx.x;
    const float* X = xyz + (size_t)b * NN * 3;
    const float cx = centers[c * 3 + 0], cy = centers[c * 3 + 1], cz = centers[c * 3 + 2];

    for (int p = lane; p < NN; p += 32) {
        float dx = X[p * 3 + 0] - cx, dy = X[p * 3 + 1] - cy, dz = X[p * 3 + 2] - cz;
        dist[p] = dx * dx + dy * dy + dz * dz;
    }
    __syncthreads();
    for (int r = 0; r < KNB; ++r) {
        float bd = 1e30f; int bi = NN;
        for (int p = lane; p < NN; p += 32) {
            float d = dist[p];
            if (d < bd || (d == bd && p < bi)) { bd = d; bi = p; }
        }
        for (int off = 16; off > 0; off >>= 1) {
            float od = __shfl_xor(bd, off, 32);
            int   oi = __shfl_xor(bi, off, 32);
            if (od < bd || (od == bd && oi < bi)) { bd = od; bi = oi; }
        }
        if (lane == 0) { nn_idx[(size_t)c * KNB + r] = bi; dist[bi] = 1e30f; }
        __syncthreads();
    }
}

// =====================================================================
// Kernel 4: fused grouping + MLP1(relu) + MLP2(relu) + maxpool over K.
// 128 threads = 4 waves, one center per wave. WMMA f32_16x16x32_f16.
//  A tile: gathered 32x96 f16 in LDS, row = [rel(3)|0|feat(64)|0x28]
//          (8B-aligned segments -> ds_store_b64 gather writes)
//  B tiles: pre-swizzled fragment blobs -> two 16B contiguous loads
//  H: 32x128 f16 staged in LDS between GEMMs
//  maxpool: per-lane max over 8 C-regs + shfl_xor(16), lanes 0..15 write
// =====================================================================
__global__ __launch_bounds__(128) void group_mlp_kernel(
        const float* __restrict__ xyz_embed,
        const float* __restrict__ features,
        const int* __restrict__ nn_idx,
        const _Float16* __restrict__ w1f,
        const _Float16* __restrict__ w2f,
        const float* __restrict__ b1,
        const float* __restrict__ b2,
        const float* __restrict__ center_embed,
        float* __restrict__ out_feat) {
    __shared__ _Float16 As[4][32 * D1P];   // 4 x 6KB
    __shared__ _Float16 Hs[4][32 * HD];    // 4 x 8KB   (total 56KB)

    const int w    = threadIdx.x >> 5;
    const int lane = threadIdx.x & 31;
    const int c    = blockIdx.x * 4 + w;
    const int b    = c >> 11;

    // ---- gather: lane r builds grouped row r = [rel(3) | 0 | feat(64) | 0]
    {
        const int idx = nn_idx[(size_t)c * KNB + lane];
        const float* em = xyz_embed + ((size_t)b * NN + idx) * 3;
        const float* fe = features  + ((size_t)b * NN + idx) * CC;
        const float cex = center_embed[c * 3 + 0];
        const float cey = center_embed[c * 3 + 1];
        const float cez = center_embed[c * 3 + 2];
        _Float16* arow = &As[w][lane * D1P];
        v4h rel = { (_Float16)(em[0] - cex), (_Float16)(em[1] - cey),
                    (_Float16)(em[2] - cez), (_Float16)0.f };
        *(v4h*)arow = rel;                              // cols 0..3 (16B aligned)
#pragma unroll
        for (int j = 0; j < CC; j += 4) {               // cols 4..67 (8B aligned)
            float4 f = *(const float4*)(fe + j);
            v4h h4 = { (_Float16)f.x, (_Float16)f.y, (_Float16)f.z, (_Float16)f.w };
            *(v4h*)(arow + 4 + j) = h4;
        }
        v4h z4 = { (_Float16)0.f, (_Float16)0.f, (_Float16)0.f, (_Float16)0.f };
#pragma unroll
        for (int j = 68; j < D1P; j += 4)               // cols 68..95
            *(v4h*)(arow + j) = z4;
    }
    __syncthreads();

    const int nlo   = lane & 15;
    const int khalf = (lane >> 4) ? 8 : 0;

    // ---- GEMM1: (32 x 96) x (96 x 128) -> H, fused bias+relu, f16 to LDS
#pragma unroll 1
    for (int nt = 0; nt < 8; ++nt) {
        const float bias = b1[nt * 16 + nlo];
#pragma unroll
        for (int mt = 0; mt < 2; ++mt) {
            v8f acc = {};
#pragma unroll
            for (int kt = 0; kt < 3; ++kt) {
                const int row = mt * 16 + nlo;
                const int kb  = kt * 32 + khalf;
                const _Float16* ap = &As[w][row * D1P + kb];
                v16h a = make_v16h(*(const v8h*)ap, *(const v8h*)(ap + 16));
                const _Float16* bp = w1f + (((kt * 8 + nt) * 512) + lane * 16);
                v16h bt = make_v16h(*(const v8h*)bp, *(const v8h*)(bp + 8));
                acc = __builtin_amdgcn_wmma_f32_16x16x32_f16(
                        false, a, false, bt, (short)0, acc, false, false);
            }
            const int mrow = mt * 16 + khalf;   // khalf==8 for lanes 16..31
#pragma unroll
            for (int v = 0; v < 8; ++v) {
                float h = acc[v] + bias;
                h = h > 0.f ? h : 0.f;
                Hs[w][(mrow + v) * HD + nt * 16 + nlo] = (_Float16)h;
            }
        }
    }
    __syncthreads();

    // ---- GEMM2: (32 x 128) x (128 x 128), fused bias+relu+maxpool(K=32)
#pragma unroll 1
    for (int nt = 0; nt < 8; ++nt) {
        const float bias = b2[nt * 16 + nlo];
        float cmax = -1e30f;
#pragma unroll
        for (int mt = 0; mt < 2; ++mt) {
            v8f acc = {};
#pragma unroll
            for (int kt = 0; kt < 4; ++kt) {
                const int row = mt * 16 + nlo;
                const int kb  = kt * 32 + khalf;
                const _Float16* ap = &Hs[w][row * HD + kb];
                v16h a = make_v16h(*(const v8h*)ap, *(const v8h*)(ap + 16));
                const _Float16* bp = w2f + (((kt * 8 + nt) * 512) + lane * 16);
                v16h bt = make_v16h(*(const v8h*)bp, *(const v8h*)(bp + 8));
                acc = __builtin_amdgcn_wmma_f32_16x16x32_f16(
                        false, a, false, bt, (short)0, acc, false, false);
            }
#pragma unroll
            for (int v = 0; v < 8; ++v) {
                float h = acc[v] + bias;
                h = h > 0.f ? h : 0.f;
                cmax = fmaxf(cmax, h);
            }
        }
        cmax = fmaxf(cmax, __shfl_xor(cmax, 16, 32));   // fold M rows 0..7 vs 8..15
        if (lane < 16) out_feat[(size_t)c * HD + nt * 16 + lane] = cmax;
    }
}

// =====================================================================
extern "C" void kernel_launch(void* const* d_in, const int* in_sizes, int n_in,
                              void* d_out, int out_size, void* d_ws, size_t ws_size,
                              hipStream_t stream) {
    const float* xyz       = (const float*)d_in[0];
    const float* xyz_embed = (const float*)d_in[1];
    const float* features  = (const float*)d_in[2];
    const float* W1        = (const float*)d_in[3];
    const float* b1        = (const float*)d_in[4];
    const float* W2        = (const float*)d_in[5];
    const float* b2        = (const float*)d_in[6];

    float* out          = (float*)d_out;
    float* out_centers  = out;                         // BM*3
    float* out_cembed   = out + (size_t)BM * 3;        // BM*3
    float* out_feat     = out + (size_t)BM * 6;        // BM*128
    float* out_ids      = out + (size_t)BM * 6 + (size_t)BM * HD; // BM

    char* ws = (char*)d_ws;
    int*      sids = (int*)ws;                                   // BM ints
    int*      nn   = (int*)(ws + (size_t)BM * 4);                // BM*32 ints
    _Float16* w1f  = (_Float16*)(ws + (size_t)BM * 4 + (size_t)BM * KNB * 4);
    _Float16* w2f  = w1f + 3 * 8 * 512;

    prep_weights_kernel<<<112, 256, 0, stream>>>(W1, W2, w1f, w2f);
    fps_kernel<<<BB, 256, 0, stream>>>(xyz, sids);
    gather_centers_kernel<<<BM / 256, 256, 0, stream>>>(xyz, xyz_embed, sids,
                                                        out_centers, out_cembed, out_ids);
    knn_kernel<<<BM, 32, 0, stream>>>(xyz, out_centers, nn);
    group_mlp_kernel<<<BM / 4, 128, 0, stream>>>(xyz_embed, features, nn,
                                                 w1f, w2f, b1, b2,
                                                 out_cembed, out_feat);
}
